// MoE_45320494907572
// MI455X (gfx1250) — matrix operand order repaired and checked by reading, not compiled, not run on hardware
//
#include <hip/hip_runtime.h>
#include <hip/hip_bf16.h>

typedef __bf16 bf16_t;
typedef __bf16 v16bf __attribute__((ext_vector_type(16)));
typedef __bf16 v8bf  __attribute__((ext_vector_type(8)));
typedef float  v8f   __attribute__((ext_vector_type(8)));
typedef float  v4f   __attribute__((ext_vector_type(4)));

#define NTOK 16384
#define DIM  1024
#define NP   256
#define NE   8
#define NH   128

__device__ __constant__ const float kClampMax = 4.6051701859880914f; // log(100)
#define EPSN 1e-12f

// ---------- helpers ----------
__device__ __forceinline__ void split_bf16(float x, bf16_t& hi, bf16_t& lo) {
  hi = (bf16_t)x;
  lo = (bf16_t)(x - (float)hi);
}

__device__ __forceinline__ v8f wmma_bf16(v16bf a, v16bf b, v8f c) {
  return __builtin_amdgcn_wmma_f32_16x16x32_bf16(false, a, false, b, (short)0, c, false, false);
}

// A fragment: 16x32 bf16. lanes<16 -> k in [kk+0..7] & [kk+16..23]; lanes>=16 -> +8.
template<int LDK>
__device__ __forceinline__ v16bf load_fragA(const bf16_t* base, int row, int kk, int lane) {
  const bf16_t* p = base + (row + (lane & 15)) * LDK + kk + ((lane >> 4) << 3);
  union { v16bf v; v8bf h[2]; } u;
  u.h[0] = *(const v8bf*)(p);
  u.h[1] = *(const v8bf*)(p + 16);
  return u.v;
}

// B fragment: 32x16 bf16, stored transposed in LDS as [n][k].
// lanes<16 -> k in [0..7] & [16..23]; lanes>=16 -> +8; n = col + lane%16.
template<int LDK>
__device__ __forceinline__ v16bf load_fragB(const bf16_t* base, int col, int lane) {
  const bf16_t* p = base + (col + (lane & 15)) * LDK + ((lane >> 4) << 3);
  union { v16bf v; v8bf h[2]; } u;
  u.h[0] = *(const v8bf*)(p);
  u.h[1] = *(const v8bf*)(p + 16);
  return u.v;
}

// ---------- K0: normalize sim_matrix, scale, zero counters ----------
__global__ void __launch_bounds__(256) moe_prep(const float* __restrict__ sim,
    const float* __restrict__ temp, float* __restrict__ smn,
    float* __restrict__ scale, int* __restrict__ cnt)
{
  __shared__ float inv[NE];
  int tid = threadIdx.x;
  if (tid < NE) {
    float s = 0.f;
    for (int p = 0; p < NP; ++p) { float v = sim[p * NE + tid]; s += v * v; }
    inv[tid] = 1.0f / fmaxf(sqrtf(s), EPSN);
    cnt[tid] = 0;
  }
  if (tid == 0) scale[0] = __expf(fminf(temp[0], kClampMax));
  __syncthreads();
  for (int i = tid; i < NP * NE; i += 256) smn[i] = sim[i] * inv[i & 7];
}

// ---------- K1: gate GEMM + l2norm + logits + top2 softmax + dispatch ----------
__global__ void __launch_bounds__(256) moe_gate(const float* __restrict__ x,
    const float* __restrict__ gate_w, const float* __restrict__ gate_b,
    const float* __restrict__ smn, const float* __restrict__ scale_p,
    int* __restrict__ cnt, int* __restrict__ list,
    float* __restrict__ gval, float* __restrict__ gsum)
{
  constexpr int LDK = 40; // 32 + 8 pad (bf16 units)
  extern __shared__ char smem[];
  bf16_t* aHi = (bf16_t*)smem;            // 32*40
  bf16_t* aLo = aHi + 32 * LDK;           // 32*40
  bf16_t* bHi = aLo + 32 * LDK;           // 256*40
  bf16_t* bLo = bHi + NP * LDK;           // 256*40
  // after the GEMM loop, overlay reduction scratch onto the B region
  float* proj = (float*)bHi;              // 32*264 f32
  float* red  = proj + 32 * 264;          // 32*8
  float* invn = red + 256;                // 32
  float* lg   = invn + 32;                // 32*8

  const int tid = threadIdx.x, lane = tid & 31, wid = tid >> 5;
  const int wr = wid >> 2, wc = wid & 3;  // 2 row-groups x 4 col-groups
  const int m0 = blockIdx.x * 32;
  const float scale = scale_p[0];

  v8f acc[4] = {};                        // 16 rows x 64 cols per wave

  for (int k0 = 0; k0 < DIM; k0 += 32) {
    { // stage A: 32 rows x 32 k
      int r = tid >> 3, c = (tid & 7) * 4;
      v4f xv = *(const v4f*)(x + (size_t)(m0 + r) * DIM + k0 + c);
#pragma unroll
      for (int j = 0; j < 4; ++j) {
        bf16_t h, l; split_bf16(xv[j], h, l);
        aHi[r * LDK + c + j] = h; aLo[r * LDK + c + j] = l;
      }
    }
    { // stage B transposed: gate_w[k0+kk][n] -> b[n][kk]
      int kk = tid >> 3, nb = (tid & 7) * 32;
      const float* gw = gate_w + (size_t)(k0 + kk) * NP + nb;
#pragma unroll
      for (int j = 0; j < 32; j += 4) {
        v4f wv = *(const v4f*)(gw + j);
#pragma unroll
        for (int q = 0; q < 4; ++q) {
          bf16_t h, l; split_bf16(wv[q], h, l);
          int n = nb + j + q;
          bHi[n * LDK + kk] = h; bLo[n * LDK + kk] = l;
        }
      }
    }
    __syncthreads();
    v16bf ah = load_fragA<LDK>(aHi, wr * 16, 0, lane);
    v16bf al = load_fragA<LDK>(aLo, wr * 16, 0, lane);
#pragma unroll
    for (int t = 0; t < 4; ++t) {
      int nc = wc * 64 + t * 16;
      v16bf bh = load_fragB<LDK>(bHi, nc, lane);
      v16bf bl = load_fragB<LDK>(bLo, nc, lane);
      acc[t] = wmma_bf16(ah, bl, acc[t]);   // hi*lo
      acc[t] = wmma_bf16(al, bh, acc[t]);   // lo*hi
      acc[t] = wmma_bf16(ah, bh, acc[t]);   // hi*hi
    }
    __syncthreads();
  }

  // write proj (+gate bias) into LDS
  {
    int rbase = wr * 16 + 8 * (lane >> 4);
#pragma unroll
    for (int t = 0; t < 4; ++t) {
      int nc = wc * 64 + t * 16 + (lane & 15);
      float bias = gate_b[nc];
#pragma unroll
      for (int j = 0; j < 8; ++j)
        proj[(rbase + j) * 264 + nc] = acc[t][j] + bias;
    }
  }
  __syncthreads();
  { // row sum of squares: 8 threads/row
    int r = tid >> 3, seg = tid & 7;
    float s = 0.f;
    for (int p = seg * 32; p < seg * 32 + 32; ++p) { float v = proj[r * 264 + p]; s += v * v; }
    red[r * 8 + seg] = s;
  }
  __syncthreads();
  if (tid < 32) {
    float s = 0.f;
    for (int j = 0; j < 8; ++j) s += red[tid * 8 + j];
    invn[tid] = 1.0f / fmaxf(sqrtf(s), EPSN);
  }
  __syncthreads();
  { // logits[r][e]
    int r = tid >> 3, e = tid & 7;
    float s = 0.f;
    for (int p = 0; p < NP; ++p) s += proj[r * 264 + p] * smn[p * NE + e];
    lg[r * NE + e] = s * invn[r] * scale;
  }
  __syncthreads();
  if ((tid & 7) == 0) { // top-2 + softmax per row
    int r = tid >> 3, token = m0 + r;
    float v0 = -1e30f, v1 = -1e30f; int i0 = 0, i1 = 0;
    for (int e = 0; e < NE; ++e) {
      float v = lg[r * NE + e];
      if (v > v0)      { v1 = v0; i1 = i0; v0 = v; i0 = e; }
      else if (v > v1) { v1 = v;  i1 = e; }
    }
    float e1 = __expf(v1 - v0);
    float g0 = 1.0f / (1.0f + e1);
    float g1 = e1 * g0;
    gval[token * 2 + 0] = g0;
    gval[token * 2 + 1] = g1;
    gsum[token] = g0 + g1;
    int p0 = atomicAdd(&cnt[i0], 1); list[i0 * NTOK + p0] = token * 2 + 0;
    int p1 = atomicAdd(&cnt[i1], 1); list[i1 * NTOK + p1] = token * 2 + 1;
  }
}

// ---------- K_res: out = x * (g0+g1) ----------
__global__ void __launch_bounds__(256) moe_resid(const float* __restrict__ x,
    const float* __restrict__ gsum, float* __restrict__ out)
{
  size_t i = ((size_t)blockIdx.x * 256 + threadIdx.x) * 8;
  float g = gsum[i >> 10]; // D = 1024
  v4f a = *(const v4f*)(x + i);
  v4f b = *(const v4f*)(x + i + 4);
#pragma unroll
  for (int q = 0; q < 4; ++q) { a[q] *= g; b[q] *= g; }
  *(v4f*)(out + i) = a;
  *(v4f*)(out + i + 4) = b;
}

// ---------- K2: down-proj per expert tile: h = relu(Xg @ w_down + b_down) ----------
__global__ void __launch_bounds__(256) moe_down(const float* __restrict__ x,
    const float* __restrict__ w_down, const float* __restrict__ b_down,
    const int* __restrict__ cnt, const int* __restrict__ list,
    float* __restrict__ h_buf)
{
  const int e = blockIdx.y;
  const int count = cnt[e];
  const int base = blockIdx.x * 32;
  if (base >= count) return;
  const int rem = min(32, count - base);

  constexpr int LDK = 40;
  extern __shared__ char smem[];
  bf16_t* aHi = (bf16_t*)smem;          // 32*40
  bf16_t* aLo = aHi + 32 * LDK;
  bf16_t* bHi = aLo + 32 * LDK;         // 128*40
  bf16_t* bLo = bHi + NH * LDK;
  int* ent = (int*)(bLo + NH * LDK);    // 32

  const int tid = threadIdx.x, lane = tid & 31, wid = tid >> 5;
  const int wr = wid >> 2, wc = wid & 3; // 2 x 4: wave = 16 rows x 32 h-cols
  if (tid < 32) ent[tid] = (tid < rem) ? list[e * NTOK + base + tid] : -1;
  __syncthreads();

  const float* wd = w_down + (size_t)e * DIM * NH;
  v8f acc[2] = {};

  for (int k0 = 0; k0 < DIM; k0 += 32) {
    { // A: gathered token rows
      int r = tid >> 3, c = (tid & 7) * 4;
      int en = ent[r];
      v4f xv = {};
      if (en >= 0) xv = *(const v4f*)(x + (size_t)(en >> 1) * DIM + k0 + c);
#pragma unroll
      for (int j = 0; j < 4; ++j) {
        bf16_t h, l; split_bf16(xv[j], h, l);
        aHi[r * LDK + c + j] = h; aLo[r * LDK + c + j] = l;
      }
    }
    { // B transposed: w_down[k][h] -> b[h][k]
      int kk = tid >> 3, hb = (tid & 7) * 16;
      const float* wp = wd + (size_t)(k0 + kk) * NH + hb;
#pragma unroll
      for (int j = 0; j < 16; j += 4) {
        v4f wv = *(const v4f*)(wp + j);
#pragma unroll
        for (int q = 0; q < 4; ++q) {
          bf16_t h, l; split_bf16(wv[q], h, l);
          int n = hb + j + q;
          bHi[n * LDK + kk] = h; bLo[n * LDK + kk] = l;
        }
      }
    }
    __syncthreads();
    v16bf ah = load_fragA<LDK>(aHi, wr * 16, 0, lane);
    v16bf al = load_fragA<LDK>(aLo, wr * 16, 0, lane);
#pragma unroll
    for (int t = 0; t < 2; ++t) {
      int nc = wc * 32 + t * 16;
      v16bf bh = load_fragB<LDK>(bHi, nc, lane);
      v16bf bl = load_fragB<LDK>(bLo, nc, lane);
      acc[t] = wmma_bf16(ah, bl, acc[t]);
      acc[t] = wmma_bf16(al, bh, acc[t]);
      acc[t] = wmma_bf16(ah, bh, acc[t]);
    }
    __syncthreads();
  }

  int rbase = wr * 16 + 8 * (lane >> 4);
#pragma unroll
  for (int t = 0; t < 2; ++t) {
    int hc = wc * 32 + t * 16 + (lane & 15);
    float bias = b_down[e * NH + hc];
#pragma unroll
    for (int j = 0; j < 8; ++j) {
      int en = ent[rbase + j];
      if (en >= 0) h_buf[(size_t)en * NH + hc] = fmaxf(acc[t][j] + bias, 0.f);
    }
  }
}

// ---------- K3: up-proj + combine (launched once per expert, stream-serialized) ----------
__global__ void __launch_bounds__(256) moe_up(const float* __restrict__ h_buf,
    const float* __restrict__ w_up, const float* __restrict__ b_up,
    const int* __restrict__ cnt, const int* __restrict__ list,
    const float* __restrict__ gval, float* __restrict__ out, int e)
{
  const int count = cnt[e];
  const int base = blockIdx.x * 32;
  if (base >= count) return;
  const int rem = min(32, count - base);

  constexpr int LDKA = 136; // 128 + 8 pad
  constexpr int LDK  = 40;
  extern __shared__ char smem[];
  bf16_t* aHi = (bf16_t*)smem;             // 32*136
  bf16_t* aLo = aHi + 32 * LDKA;
  bf16_t* bHi = aLo + 32 * LDKA;           // 256*40
  bf16_t* bLo = bHi + 256 * LDK;
  int*   ent = (int*)(bLo + 256 * LDK);    // 32
  float* gv  = (float*)(ent + 32);         // 32

  const int tid = threadIdx.x, lane = tid & 31, wid = tid >> 5;
  const int wr = wid >> 2, wc = wid & 3;   // wave = 16 rows x 64 cols (of 256 chunk)
  if (tid < 32) {
    int en = (tid < rem) ? list[e * NTOK + base + tid] : -1;
    ent[tid] = en;
    gv[tid] = (en >= 0) ? gval[en] : 0.f;
  }
  __syncthreads();

  { // stage A = g * h (full K=128) split into hi/lo
    int r = tid >> 3, c = (tid & 7) * 16;
    int en = ent[r]; float g = gv[r];
#pragma unroll
    for (int j = 0; j < 16; j += 4) {
      v4f hv = {};
      if (en >= 0) hv = *(const v4f*)(h_buf + (size_t)en * NH + c + j);
#pragma unroll
      for (int q = 0; q < 4; ++q) {
        bf16_t h, l; split_bf16(hv[q] * g, h, l);
        aHi[r * LDKA + c + j + q] = h; aLo[r * LDKA + c + j + q] = l;
      }
    }
  }
  __syncthreads();

  const float* wu = w_up + (size_t)e * NH * DIM;
  for (int n0 = 0; n0 < DIM; n0 += 256) {
    v8f acc[4] = {};
    for (int k0 = 0; k0 < NH; k0 += 32) {
      { // B chunk transposed: w_up[k][n0+n] -> b[n][kk]
        int kk = tid >> 3, nb = (tid & 7) * 32;
        const float* wp = wu + (size_t)(k0 + kk) * DIM + n0 + nb;
#pragma unroll
        for (int j = 0; j < 32; j += 4) {
          v4f wv = *(const v4f*)(wp + j);
#pragma unroll
          for (int q = 0; q < 4; ++q) {
            bf16_t h, l; split_bf16(wv[q], h, l);
            int n = nb + j + q;
            bHi[n * LDK + kk] = h; bLo[n * LDK + kk] = l;
          }
        }
      }
      __syncthreads();
      v16bf ah = load_fragA<LDKA>(aHi, wr * 16, k0, lane);
      v16bf al = load_fragA<LDKA>(aLo, wr * 16, k0, lane);
#pragma unroll
      for (int t = 0; t < 4; ++t) {
        int nc = wc * 64 + t * 16;
        v16bf bh = load_fragB<LDK>(bHi, nc, lane);
        v16bf bl = load_fragB<LDK>(bLo, nc, lane);
        acc[t] = wmma_bf16(ah, bl, acc[t]);
        acc[t] = wmma_bf16(al, bh, acc[t]);
        acc[t] = wmma_bf16(ah, bh, acc[t]);
      }
      __syncthreads();
    }
    // combine: out[token][col] += acc + g * b_up[e][col]
    int rbase = wr * 16 + 8 * (lane >> 4);
#pragma unroll
    for (int t = 0; t < 4; ++t) {
      int col = n0 + wc * 64 + t * 16 + (lane & 15);
      float bb = b_up[(size_t)e * DIM + col];
#pragma unroll
      for (int j = 0; j < 8; ++j) {
        int r = rbase + j;
        int en = ent[r];
        if (en >= 0) {
          size_t o = (size_t)(en >> 1) * DIM + col;
          out[o] += acc[t][j] + gv[r] * bb;
        }
      }
    }
  }
}

// ---------- host ----------
extern "C" void kernel_launch(void* const* d_in, const int* in_sizes, int n_in,
                              void* d_out, int out_size, void* d_ws, size_t ws_size,
                              hipStream_t stream) {
  (void)in_sizes; (void)n_in; (void)out_size; (void)ws_size;
  const float* x      = (const float*)d_in[0];
  const float* gate_w = (const float*)d_in[1];
  const float* gate_b = (const float*)d_in[2];
  const float* sim    = (const float*)d_in[3];
  const float* temp   = (const float*)d_in[4];
  const float* w_down = (const float*)d_in[5];
  const float* b_down = (const float*)d_in[6];
  const float* w_up   = (const float*)d_in[7];
  const float* b_up   = (const float*)d_in[8];
  float* out = (float*)d_out;

  // workspace layout
  char* ws = (char*)d_ws;
  float* smn   = (float*)(ws);             // 2048 f32
  float* scale = (float*)(ws + 8192);      // 1 f32
  int*   cnt   = (int*)  (ws + 8448);      // 8 int
  float* gval  = (float*)(ws + 8704);      // N*2 f32
  float* gsum  = (float*)(ws + 139776);    // N f32
  int*   list  = (int*)  (ws + 205312);    // E*N int
  float* hbuf  = (float*)(ws + 729600);    // N*2*128 f32 (16 MB)

  constexpr size_t lds_gate = 2 * 32 * 40 * 2 + 2 * 256 * 40 * 2;          // 46080 B
  constexpr size_t lds_down = 2 * 32 * 40 * 2 + 2 * 128 * 40 * 2 + 128;    // 25728 B
  constexpr size_t lds_up   = 2 * 32 * 136 * 2 + 2 * 256 * 40 * 2 + 256;   // 58624 B

  moe_prep<<<1, 256, 0, stream>>>(sim, temp, smn, scale, cnt);
  moe_gate<<<NTOK / 32, 256, lds_gate, stream>>>(x, gate_w, gate_b, smn, scale,
                                                 cnt, list, gval, gsum);
  moe_resid<<<(NTOK * DIM) / (256 * 8), 256, 0, stream>>>(x, gsum, out);
  moe_down<<<dim3(NTOK / 32, NE), 256, lds_down, stream>>>(x, w_down, b_down,
                                                           cnt, list, hbuf);
  for (int e = 0; e < NE; ++e)
    moe_up<<<NTOK / 32, 256, lds_up, stream>>>(hbuf, w_up, b_up, cnt, list,
                                               gval, out, e);
}
